// OneBitLinear_62414464745557
// MI455X (gfx1250) — matrix operand order repaired and checked by reading, not compiled, not run on hardware
//
#include <hip/hip_runtime.h>
#include <stdint.h>

#define IN_F   4096
#define OUT_F  4096
#define M_TOT  8192

#define TILE_M 256
#define TILE_N 128
#define TILE_K 64
#define NKITER (IN_F / TILE_K)   // 64
#define LDSW   72                // LDS row stride in bf16 elements (144 B): conflict-free b128 reads
#define NTHREADS 512             // 16 waves

typedef __attribute__((ext_vector_type(16))) __bf16        v16bf;
typedef __attribute__((ext_vector_type(8)))  float         v8f;
typedef __attribute__((ext_vector_type(4)))  unsigned int  u32x4;
typedef __attribute__((ext_vector_type(8)))  unsigned int  u32x8;

// bf16(x0*s0) in [15:0], bf16(x1*s1) in [31:16] -- VOP3P mix ops:
// fuse multiply + f32->bf16 RNE convert + halfword pack.
__device__ __forceinline__ unsigned int fma_pack_bf16(float x0, float s0, float x1, float s1) {
    unsigned int d;
    asm("v_fma_mixlo_bf16 %0, %1, %2, 0\n\t"
        "v_fma_mixhi_bf16 %0, %3, %4, 0"
        : "=&v"(d)
        : "v"(x0), "v"(s0), "v"(x1), "v"(s1));
    return d;
}

// one byte of 8 MSB-first sign bits -> 4 packed bf16x2 (bit 1 -> +1.0, bit 0 -> -1.0)
__device__ __forceinline__ u32x4 unpack8signs(unsigned int wb) {
    const unsigned int nw = ~wb;
    u32x4 q;
#pragma unroll
    for (int i = 0; i < 4; i++) {
        q[i] = 0x3F803F80u
             | ((nw << (8 + 2 * i)) & 0x8000u)          // bit (7-2i) -> sign of lo half
             | ((nw << (25 + 2 * i)) & 0x80000000u);    // bit (6-2i) -> sign of hi half
    }
    return q;
}

// workgroup barrier that drains ONLY dscnt (LDS ops) -- unlike __syncthreads(),
// it does NOT wait for in-flight global loads, so the next-tile register
// prefetch overlaps with the WMMA section.
__device__ __forceinline__ void lds_barrier() {
    asm volatile("s_wait_dscnt 0x0\n\t"
                 "s_barrier_signal -1\n\t"
                 "s_barrier_wait -1" ::: "memory");
}

__global__ __launch_bounds__(NTHREADS)
void onebit_gemm_bf16_wmma(const float* __restrict__ x,
                           const float* __restrict__ a,
                           const float* __restrict__ b,
                           const int*   __restrict__ sign_packed,
                           const float* __restrict__ bias,
                           float* __restrict__ out)
{
    // double-buffered tiles
    __shared__ __align__(16) unsigned short sA[2][TILE_M * LDSW]; // (x*b) bf16, [M][K]
    __shared__ __align__(16) unsigned short sW[2][TILE_N * LDSW]; // +-1   bf16, [N][K]

    const int t    = threadIdx.x;
    const int lane = t & 31;
    const int wv   = t >> 5;               // 0..15

    const int m0 = blockIdx.y * TILE_M;
    const int n0 = blockIdx.x * TILE_N;

    // A staging: 2 threads per row (256 rows), 32 K-elements each
    const int srow  = t >> 1;              // 0..255
    const int khalf = (t & 1) << 5;        // 0 or 32
    // W staging: 4 threads per row (128 rows), 16 K-elements each
    const int wrow = t >> 2;               // 0..127
    const int wk   = (t & 3) << 4;         // 0,16,32,48

    // wave tile: 4 (M) x 4 (N) waves, each 64M x 32N
    const int wm  = (wv & 3) * 64;
    const int wn  = (wv >> 2) * 32;
    const int lr  = lane & 15;
    const int sel = lane >> 4;             // A: 0 -> K{0-7,16-23}; 1 -> K{8-15,24-31}

    const float* xrow = x + (size_t)(m0 + srow) * IN_F + khalf;
    const int*   wsp  = sign_packed + (size_t)(n0 + wrow) * (IN_F / 8) + ((t & 3) << 1);

    v8f acc[4][2];
#pragma unroll
    for (int tm = 0; tm < 4; tm++)
#pragma unroll
        for (int tn = 0; tn < 2; tn++)
            acc[tm][tn] = (v8f){0.f, 0.f, 0.f, 0.f, 0.f, 0.f, 0.f, 0.f};

    // ---------------- register prefetch of tile 0 ----------------
    float4 px[8];
    int2   pw;
    {
        const float4* xp = (const float4*)xrow;
#pragma unroll
        for (int i = 0; i < 8; i++) px[i] = xp[i];
        pw = *(const int2*)wsp;
    }

    for (int kt = 0; kt < NKITER; ++kt) {
        const int k0  = kt * TILE_K;
        const int buf = kt & 1;

        // ---------- stage prefetched tile into LDS[buf] ----------
        {
            // A = (x * b) -> bf16 via v_fma_mixlo/hi_bf16
            const float4* bp = (const float4*)(b + k0 + khalf);
            u32x4* Ad = (u32x4*)&sA[buf][srow * LDSW + khalf];
#pragma unroll
            for (int c = 0; c < 4; c++) {
                float4 b0 = bp[2 * c], b1 = bp[2 * c + 1];
                float4 x0 = px[2 * c], x1 = px[2 * c + 1];
                u32x4 p;
                p[0] = fma_pack_bf16(x0.x, b0.x, x0.y, b0.y);
                p[1] = fma_pack_bf16(x0.z, b0.z, x0.w, b0.w);
                p[2] = fma_pack_bf16(x1.x, b1.x, x1.y, b1.y);
                p[3] = fma_pack_bf16(x1.z, b1.z, x1.w, b1.w);
                Ad[c] = p;
            }
            // W: sign bits -> +-1 bf16 (each int32 holds one byte, MSB-first)
            u32x4* Wd = (u32x4*)&sW[buf][wrow * LDSW + wk];
            Wd[0] = unpack8signs((unsigned int)pw.x & 0xffu);
            Wd[1] = unpack8signs((unsigned int)pw.y & 0xffu);
        }

        // ---------- issue global loads for next tile (overlap with WMMA section) ----------
        if (kt + 1 < NKITER) {
            const int kn = (kt + 1) * TILE_K;
            const float4* xp = (const float4*)(xrow + kn);
#pragma unroll
            for (int i = 0; i < 8; i++) px[i] = xp[i];
            pw = *(const int2*)(wsp + (kn >> 3));
            if (kt + 2 < NKITER)  // keep the x stream warm in L2
                __builtin_prefetch(xrow + (kt + 2) * TILE_K, 0, 1);
        }

        lds_barrier();   // drains dscnt only; global prefetch stays in flight

        // ---------- fragments (ISA 7.12.2 layouts) & 16 WMMAs ----------
        v16bf bfr[2][2];   // [ks][tn]
#pragma unroll
        for (int ks = 0; ks < 2; ks++)
#pragma unroll
            for (int tn = 0; tn < 2; tn++) {
                // B 32x16: lane<16 K{0-7,16-23} of column lr; lane>=16 K{8-15,24-31}
                const u32x4* p = (const u32x4*)
                    &sW[buf][(wn + tn * 16 + lr) * LDSW + ks * 32 + sel * 8];
                u32x4 lo = p[0], hi = p[2];
                u32x8 r;
#pragma unroll
                for (int j = 0; j < 4; j++) { r[j] = lo[j]; r[4 + j] = hi[j]; }
                bfr[ks][tn] = __builtin_bit_cast(v16bf, r);
            }

#pragma unroll
        for (int tm = 0; tm < 4; tm++) {
            v16bf afr[2];
#pragma unroll
            for (int ks = 0; ks < 2; ks++) {
                // A 16x32: lane<16 K{0-7,16-23}; lane>=16 K{8-15,24-31}
                const u32x4* p = (const u32x4*)
                    &sA[buf][(wm + tm * 16 + lr) * LDSW + ks * 32 + sel * 8];
                u32x4 lo = p[0], hi = p[2];
                u32x8 r;
#pragma unroll
                for (int j = 0; j < 4; j++) { r[j] = lo[j]; r[4 + j] = hi[j]; }
                afr[ks] = __builtin_bit_cast(v16bf, r);
            }
#pragma unroll
            for (int ks = 0; ks < 2; ks++) {
                acc[tm][0] = __builtin_amdgcn_wmma_f32_16x16x32_bf16(
                    false, afr[ks], false, bfr[ks][0], (short)0, acc[tm][0], false, false);
                acc[tm][1] = __builtin_amdgcn_wmma_f32_16x16x32_bf16(
                    false, afr[ks], false, bfr[ks][1], (short)0, acc[tm][1], false, false);
            }
        }
    }

    // ---------- epilogue: out = a[n]*acc + bias[n], nontemporal stores ----------
#pragma unroll
    for (int tn = 0; tn < 2; tn++) {
        const int n  = n0 + wn + tn * 16 + lr;
        const float av = a[n];
        const float bv = bias[n];
#pragma unroll
        for (int tm = 0; tm < 4; tm++) {
            const int mbase = m0 + wm + tm * 16 + sel * 8;  // lanes 16-31 hold rows +8
#pragma unroll
            for (int r = 0; r < 8; r++) {
                float v = av * acc[tm][tn][r] + bv;
                __builtin_nontemporal_store(v, out + (size_t)(mbase + r) * OUT_F + n);
            }
        }
    }
}

extern "C" void kernel_launch(void* const* d_in, const int* in_sizes, int n_in,
                              void* d_out, int out_size, void* d_ws, size_t ws_size,
                              hipStream_t stream) {
    (void)in_sizes; (void)n_in; (void)out_size; (void)d_ws; (void)ws_size;
    const float* x           = (const float*)d_in[0];
    const float* a           = (const float*)d_in[1];
    const float* b           = (const float*)d_in[2];
    const int*   sign_packed = (const int*)d_in[3];
    const float* bias        = (const float*)d_in[4];
    float*       out         = (float*)d_out;

    dim3 grid(OUT_F / TILE_N, M_TOT / TILE_M);   // (32, 32)
    onebit_gemm_bf16_wmma<<<grid, dim3(NTHREADS), 0, stream>>>(x, a, b, sign_packed, bias, out);
}
